// DecoupledAstormer_72576357368449
// MI455X (gfx1250) — compile-verified, matrix operand order checked by reading, not run on hardware
//
#include <hip/hip_runtime.h>

#define HIDN 512
#define NH 8
#define HD 64
#define NLAYERS 2
#define BB 2
#define TT 512
#define SSK 512
#define RN_ 13
#define RPAD 16
#define INV_SCALE 0.125f

typedef __attribute__((ext_vector_type(16))) __bf16 v16bf;
typedef __attribute__((ext_vector_type(8)))  float  v8f;
typedef __attribute__((ext_vector_type(2)))  float  v2f;
typedef __attribute__((ext_vector_type(2)))  __bf16 v2bf;

// ---------- helpers ----------

// pack two floats into one u32 of bf16 pair via native convert (v_cvt_pk_bf16_f32)
__device__ __forceinline__ unsigned pkbf(float a, float b) {
  v2f f; f.x = a; f.y = b;
  v2bf h = __builtin_convertvector(f, v2bf);
  unsigned u; __builtin_memcpy(&u, &h, 4);
  return u;
}

// Load a WMMA 16x32 bf16 A/B fragment from LDS.
// Layout (CDNA5 ISA 7.12.2, 16-bit A 16x32): lane l holds row m=l&15,
// K = {(l>>4)*8 + 0..7, 16 + (l>>4)*8 + 0..7} -> two contiguous 16B chunks.
__device__ __forceinline__ v16bf frag_ld(const __bf16* base, int lane, int stride) {
  const int m  = lane & 15;
  const int kb = (lane >> 4) * 8;
  const __bf16* row = base + m * stride + kb;
  v16bf r;
#pragma unroll
  for (int i = 0; i < 8; ++i) r[i] = row[i];
#pragma unroll
  for (int i = 0; i < 8; ++i) r[8 + i] = row[16 + i];
  return r;
}

__device__ __forceinline__ v8f wmma_bf16(v16bf a, v16bf b, v8f c) {
  return __builtin_amdgcn_wmma_f32_16x16x32_bf16(false, a, false, b, (short)0, c,
                                                 false, false);
}

// ---------- GEMM: C[M,N] = act(A[M,K] @ W[K,N] + bias) ----------
// block = 128 threads (4 waves), tile 64x64, K-step 32.
template <int ACT>
__global__ __launch_bounds__(128) void gemm_kernel(
    const float* __restrict__ A, const float* __restrict__ W,
    const float* __restrict__ bias, float* __restrict__ C,
    int M, int N, int K) {
  __shared__ unsigned AsU[64][16];  // A tile, bf16 row-major [64][32]
  __shared__ unsigned BtU[64][16];  // W tile transposed: bf16 Bt[n][k] = W[k][n]
  const int tid  = threadIdx.x;
  const int wave = tid >> 5, lane = tid & 31;
  const int m0 = blockIdx.y * 64, n0 = blockIdx.x * 64;
  const int wm = (wave >> 1) * 32, wn = (wave & 1) * 32;
  v8f acc[2][2] = {};
  for (int k0 = 0; k0 < K; k0 += 32) {
    // A: float4 loads, packed b64 LDS stores (4 unconditional iterations)
#pragma unroll
    for (int it = 0; it < 4; ++it) {
      const int i = it * 512 + tid * 4;
      const int r = i >> 5, c = i & 31;  // c multiple of 4
      const float4 f = *(const float4*)&A[(size_t)(m0 + r) * K + (k0 + c)];
      uint2 p; p.x = pkbf(f.x, f.y); p.y = pkbf(f.z, f.w);
      *(uint2*)&AsU[r][c >> 1] = p;
    }
    // W: pairs of k rows, float4 along n, packed (k,k+1) b32 stores (2 iters)
#pragma unroll
    for (int it = 0; it < 2; ++it) {
      const int i = it * 128 + tid;
      const int kp = i & 15;            // k pair -> k = 2*kp
      const int n4 = (i >> 4) * 4;      // n base
      const float4 r0 = *(const float4*)&W[(size_t)(k0 + 2 * kp + 0) * N + (n0 + n4)];
      const float4 r1 = *(const float4*)&W[(size_t)(k0 + 2 * kp + 1) * N + (n0 + n4)];
      BtU[n4 + 0][kp] = pkbf(r0.x, r1.x);
      BtU[n4 + 1][kp] = pkbf(r0.y, r1.y);
      BtU[n4 + 2][kp] = pkbf(r0.z, r1.z);
      BtU[n4 + 3][kp] = pkbf(r0.w, r1.w);
    }
    __syncthreads();
    const v16bf a0 = frag_ld((const __bf16*)&AsU[wm][0],      lane, 32);
    const v16bf a1 = frag_ld((const __bf16*)&AsU[wm + 16][0], lane, 32);
    const v16bf b0 = frag_ld((const __bf16*)&BtU[wn][0],      lane, 32);
    const v16bf b1 = frag_ld((const __bf16*)&BtU[wn + 16][0], lane, 32);
    acc[0][0] = wmma_bf16(a0, b0, acc[0][0]);
    acc[0][1] = wmma_bf16(a0, b1, acc[0][1]);
    acc[1][0] = wmma_bf16(a1, b0, acc[1][0]);
    acc[1][1] = wmma_bf16(a1, b1, acc[1][1]);
    __syncthreads();
  }
  const int n = lane & 15, mr = (lane >> 4) * 8;
#pragma unroll
  for (int mi = 0; mi < 2; ++mi)
#pragma unroll
    for (int ni = 0; ni < 2; ++ni) {
      const int gn = n0 + wn + ni * 16 + n;
      const float bv = bias ? bias[gn] : 0.f;
#pragma unroll
      for (int v = 0; v < 8; ++v) {
        const int gm = m0 + wm + mi * 16 + mr + v;
        float val = acc[mi][ni][v] + bv;
        if (ACT == 1) val = fmaxf(val, 0.f);
        C[(size_t)gm * N + gn] = val;
      }
    }
}

// ---------- P[b,t,h,r] = Q[b,t,h,:] . embed_k[r,:]  (RPAD-padded r) ----------
__global__ __launch_bounds__(256) void relp_kernel(
    const float* __restrict__ Q, const float* __restrict__ ek,
    float* __restrict__ P, int Tq) {
  const int idx = blockIdx.x * 256 + threadIdx.x;
  if (idx >= BB * Tq * NH * RPAD) return;
  const int r  = idx & (RPAD - 1);
  const int h  = (idx >> 4) & (NH - 1);
  const int bt = idx >> 7;  // RPAD*NH == 128
  float acc = 0.f;
  if (r < RN_) {
    const float* q = Q + (size_t)bt * HIDN + h * HD;
    const float* e = ek + r * HD;
#pragma unroll
    for (int d = 0; d < HD; d += 4) {
      const float4 qa = *(const float4*)&q[d];
      const float4 ea = *(const float4*)&e[d];
      acc += qa.x * ea.x + qa.y * ea.y + qa.z * ea.z + qa.w * ea.w;
    }
  }
  P[idx] = acc;
}

// ---------- Attention. MODE 0: rel+causal (ids!=PAD & s<=t). MODE 1: enc_mask ----------
// grid = (Tq/16, NH, BB), block = 128 (4 waves). One block = 16 query rows of one head.
template <int MODE>
__global__ __launch_bounds__(128) void attn_kernel(
    const float* __restrict__ Q, const float* __restrict__ Km,
    const float* __restrict__ Vm, const float* __restrict__ P,
    const int* __restrict__ ids, const float* __restrict__ embed_v,
    const unsigned char* __restrict__ emask, float* __restrict__ O,
    int Tq, int Sk) {
  __shared__ float    sc[16][SSK];     // score / prob strip (32 KB)
  __shared__ unsigned qsU[16][32];     // Q tile bf16 [16][64] (2 KB)
  __shared__ unsigned ktU[64][32];     // K rows bf16 [64][64] (8 KB)
  __shared__ unsigned vtU[64][16];     // V^T bf16 [64][32]: vt[d][s] (4 KB)
  __shared__ float    bins[16][RPAD];  // rel-v mass bins (1 KB)
  const int tid = threadIdx.x, wave = tid >> 5, lane = tid & 31;
  const int t0 = blockIdx.x * 16, h = blockIdx.y, b = blockIdx.z;
  const int sEnd = (MODE == 0) ? ((t0 + 16 < Sk) ? t0 + 16 : Sk) : Sk;

  // stage Q tile (float4 loads, packed b64 stores; 2 unconditional iters)
#pragma unroll
  for (int it = 0; it < 2; ++it) {
    const int i = it * 512 + tid * 4;
    const int r = i >> 6, d = i & 63;
    const float4 f = *(const float4*)&Q[(size_t)(b * Tq + t0 + r) * HIDN + h * HD + d];
    uint2 p; p.x = pkbf(f.x, f.y); p.y = pkbf(f.z, f.w);
    *(uint2*)&qsU[r][d >> 1] = p;
  }
  __syncthreads();

  // ---- scores: e = Q K^T, 64 keys per iteration, wave handles 16 columns ----
  const int scol = wave * 16;
  for (int s0 = 0; s0 < sEnd; s0 += 64) {
    // stage 64 K rows (8 unconditional iters, float4)
#pragma unroll
    for (int it = 0; it < 8; ++it) {
      const int i = it * 512 + tid * 4;
      const int s = i >> 6, d = i & 63;
      int sg = s0 + s; if (sg > Sk - 1) sg = Sk - 1;
      const float4 f = *(const float4*)&Km[(size_t)(b * Sk + sg) * HIDN + h * HD + d];
      uint2 p; p.x = pkbf(f.x, f.y); p.y = pkbf(f.z, f.w);
      *(uint2*)&ktU[s][d >> 1] = p;
    }
    __syncthreads();
    v8f e = {};
    e = wmma_bf16(frag_ld((const __bf16*)&qsU[0][0],  lane, 64),
                  frag_ld((const __bf16*)&ktU[scol][0],  lane, 64), e);
    e = wmma_bf16(frag_ld((const __bf16*)&qsU[0][16], lane, 64),
                  frag_ld((const __bf16*)&ktU[scol][16], lane, 64), e);
    const int n = lane & 15, mr = (lane >> 4) * 8;
#pragma unroll
    for (int v = 0; v < 8; ++v) {
      const int t = t0 + mr + v;
      const int s = s0 + scol + n;
      float val = e[v];
      if (MODE == 0) {
        const int sg = (s < Sk) ? s : Sk - 1;
        const int id = ids[(size_t)(b * Tq + t) * Sk + sg];
        val = (val + P[((size_t)(b * Tq + t) * NH + h) * RPAD + id]) * INV_SCALE;
        val = (id != 0 && s <= t) ? val : -1e10f;
      } else {
        const int sg = (s < Sk) ? s : Sk - 1;
        val *= INV_SCALE;
        val = (emask[b * Sk + sg] != 0) ? val : -1e10f;
      }
      sc[mr + v][s] = val;
    }
    __syncthreads();
  }

  // ---- softmax per row + rel-v bins ----
  const int avHi = (sEnd + 31) & ~31;
  if (tid < 16) {
    const int t = t0 + tid;
    int lim = (MODE == 0) ? (t + 1) : sEnd;
    if (lim > sEnd) lim = sEnd;
    float mx = -3e38f;
    for (int s = 0; s < lim; ++s) mx = fmaxf(mx, sc[tid][s]);
    float sum = 0.f;
    for (int s = 0; s < lim; ++s) { const float p = __expf(sc[tid][s] - mx); sc[tid][s] = p; sum += p; }
    const float inv = 1.f / sum;
#pragma unroll
    for (int r = 0; r < RPAD; ++r) bins[tid][r] = 0.f;
    for (int s = 0; s < lim; ++s) {
      const float p = sc[tid][s] * inv;
      sc[tid][s] = p;
      if (MODE == 0) bins[tid][ids[(size_t)(b * Tq + t) * Sk + s]] += p;
    }
    for (int s = lim; s < avHi; ++s) sc[tid][s] = 0.f;
  }
  __syncthreads();

  // ---- O = A V : wave handles 16 of 64 head dims ----
  const int dcol = wave * 16;
  v8f oa = {};
  for (int s0 = 0; s0 < sEnd; s0 += 32) {
    // stage V^T: pairs of s rows, float4 along d, packed (s,s+1) b32 stores
#pragma unroll
    for (int it = 0; it < 2; ++it) {
      const int i = it * 128 + tid;
      const int sp = i & 15;          // s pair -> s = 2*sp
      const int d4 = (i >> 4) * 4;
      int s_a = s0 + 2 * sp, s_b = s_a + 1;
      if (s_a > Sk - 1) s_a = Sk - 1;
      if (s_b > Sk - 1) s_b = Sk - 1;
      const float4 r0 = *(const float4*)&Vm[(size_t)(b * Sk + s_a) * HIDN + h * HD + d4];
      const float4 r1 = *(const float4*)&Vm[(size_t)(b * Sk + s_b) * HIDN + h * HD + d4];
      vtU[d4 + 0][sp] = pkbf(r0.x, r1.x);
      vtU[d4 + 1][sp] = pkbf(r0.y, r1.y);
      vtU[d4 + 2][sp] = pkbf(r0.z, r1.z);
      vtU[d4 + 3][sp] = pkbf(r0.w, r1.w);
    }
    __syncthreads();
    // probs fragment straight out of LDS in A layout (packed bf16 words)
    v16bf ap;
    {
      const int m = lane & 15, kb = (lane >> 4) * 8;
      const float* row = &sc[m][s0 + kb];
      unsigned aw[8];
#pragma unroll
      for (int i = 0; i < 4; ++i) aw[i] = pkbf(row[2 * i], row[2 * i + 1]);
#pragma unroll
      for (int i = 0; i < 4; ++i) aw[4 + i] = pkbf(row[16 + 2 * i], row[16 + 2 * i + 1]);
      __builtin_memcpy(&ap, aw, 32);
    }
    oa = wmma_bf16(ap, frag_ld((const __bf16*)&vtU[dcol][0], lane, 32), oa);
    __syncthreads();
  }

  // ---- write out (+ bins @ embed_v), staging frags through LDS ----
  float(*ob)[64] = (float(*)[64])sc;  // reuse score LDS
  {
    const int n = lane & 15, mr = (lane >> 4) * 8;
#pragma unroll
    for (int v = 0; v < 8; ++v) ob[mr + v][dcol + n] = oa[v];
  }
  __syncthreads();
#pragma unroll
  for (int it = 0; it < 2; ++it) {
    const int i = it * 512 + tid * 4;
    const int r = i >> 6, d = i & 63;
    float4 val = *(const float4*)&ob[r][d];
    if (MODE == 0) {
#pragma unroll
      for (int rr = 0; rr < RN_; ++rr) {
        const float w = bins[r][rr];
        const float4 ev = *(const float4*)&embed_v[rr * HD + d];
        val.x += w * ev.x; val.y += w * ev.y; val.z += w * ev.z; val.w += w * ev.w;
      }
    }
    *(float4*)&O[(size_t)(b * Tq + t0 + r) * HIDN + h * HD + d] = val;
  }
}

// ---------- out = LayerNorm(x + y) * g + b ; one block per row of 512 ----------
__global__ __launch_bounds__(128) void add_ln_kernel(
    const float* __restrict__ x, const float* __restrict__ y,
    const float* __restrict__ g, const float* __restrict__ be,
    float* __restrict__ out) {
  __shared__ float red[128];
  const int row = blockIdx.x, t = threadIdx.x;
  const size_t base = (size_t)row * HIDN;
  const int c = t * 4;
  const float4 xv = *(const float4*)&x[base + c];
  const float4 yv = *(const float4*)&y[base + c];
  float v[4] = { xv.x + yv.x, xv.y + yv.y, xv.z + yv.z, xv.w + yv.w };
  float s = v[0] + v[1] + v[2] + v[3];
  red[t] = s; __syncthreads();
  for (int o = 64; o > 0; o >>= 1) { if (t < o) red[t] += red[t + o]; __syncthreads(); }
  const float mean = red[0] * (1.f / HIDN);
  __syncthreads();
  float sq = 0.f;
#pragma unroll
  for (int j = 0; j < 4; ++j) { const float d = v[j] - mean; sq += d * d; }
  red[t] = sq; __syncthreads();
  for (int o = 64; o > 0; o >>= 1) { if (t < o) red[t] += red[t + o]; __syncthreads(); }
  const float inv = rsqrtf(red[0] * (1.f / HIDN) + 1e-5f);
  const float4 gv = *(const float4*)&g[c];
  const float4 bv = *(const float4*)&be[c];
  float4 ov;
  ov.x = (v[0] - mean) * inv * gv.x + bv.x;
  ov.y = (v[1] - mean) * inv * gv.y + bv.y;
  ov.z = (v[2] - mean) * inv * gv.z + bv.z;
  ov.w = (v[3] - mean) * inv * gv.w + bv.w;
  *(float4*)&out[base + c] = ov;
}

// ---------- host ----------
extern "C" void kernel_launch(void* const* d_in, const int* in_sizes, int n_in,
                              void* d_out, int out_size, void* d_ws, size_t ws_size,
                              hipStream_t stream) {
  const float* x_in    = (const float*)d_in[0];
  const float* prev_in = (const float*)d_in[1];
  const float* kv_in   = (const float*)d_in[2];
  // d_in[3] = rel_ids: only feeds the dead self-attention branch -> unused.
  const int* cross_ids = (const int*)d_in[4];
  const unsigned char* emask = (const unsigned char*)d_in[5];
  const float* embed_k = (const float*)d_in[6];
  const float* embed_v = (const float*)d_in[7];

  float* ws = (float*)d_ws;
  const size_t AROW = (size_t)BB * TT;   // 1024
  const size_t ASZ  = AROW * HIDN;       // 524288
  float* xA = ws; ws += ASZ;
  float* xB = ws; ws += ASZ;
  float* qb = ws; ws += ASZ;
  float* kb = ws; ws += ASZ;
  float* vb = ws; ws += ASZ;
  float* ob = ws; ws += ASZ;
  float* yb = ws; ws += ASZ;
  float* pb = ws; ws += (size_t)BB * TT * NH * RPAD;
  float* hb = ws;                        // 1024 x 2048

  const dim3 blk(128);
  const dim3 g512(HIDN / 64, AROW / 64);
  const dim3 g2048((4 * HIDN) / 64, AROW / 64);
  const dim3 attnG(TT / 16, NH, BB);
  const dim3 lnG((unsigned)AROW);

  const float* x = x_in;
  for (int L = 0; L < NLAYERS; ++L) {
    const int base = 8 + L * 30;  // self(0..7) tgt(8..15) cross(16..23) ffn(24..29)
    auto p = [&](int i) { return (const float*)d_in[base + i]; };

    // ---- tgt rel-attention: Q from x, K/V from prev ----
    gemm_kernel<0><<<g512, blk, 0, stream>>>(x,       p(8 + 0), p(8 + 1), qb, (int)AROW, HIDN, HIDN);
    gemm_kernel<0><<<g512, blk, 0, stream>>>(prev_in, p(8 + 2), nullptr,  kb, (int)AROW, HIDN, HIDN);
    gemm_kernel<0><<<g512, blk, 0, stream>>>(prev_in, p(8 + 3), nullptr,  vb, (int)AROW, HIDN, HIDN);
    relp_kernel<<<dim3((BB * TT * NH * RPAD) / 256), dim3(256), 0, stream>>>(qb, embed_k, pb, TT);
    attn_kernel<0><<<attnG, blk, 0, stream>>>(qb, kb, vb, pb, cross_ids, embed_v, nullptr, ob, TT, TT);
    gemm_kernel<0><<<g512, blk, 0, stream>>>(ob, p(8 + 4), p(8 + 5), yb, (int)AROW, HIDN, HIDN);
    add_ln_kernel<<<lnG, blk, 0, stream>>>(x, yb, p(8 + 6), p(8 + 7), xA);

    // ---- cross attention: Q from xA, K/V from kv ----
    gemm_kernel<0><<<g512, blk, 0, stream>>>(xA,    p(16 + 0), p(16 + 1), qb, (int)AROW, HIDN, HIDN);
    gemm_kernel<0><<<g512, blk, 0, stream>>>(kv_in, p(16 + 2), nullptr,   kb, (int)AROW, HIDN, HIDN);
    gemm_kernel<0><<<g512, blk, 0, stream>>>(kv_in, p(16 + 3), nullptr,   vb, (int)AROW, HIDN, HIDN);
    attn_kernel<1><<<attnG, blk, 0, stream>>>(qb, kb, vb, nullptr, nullptr, nullptr, emask, ob, TT, SSK);
    gemm_kernel<0><<<g512, blk, 0, stream>>>(ob, p(16 + 4), p(16 + 5), yb, (int)AROW, HIDN, HIDN);
    add_ln_kernel<<<lnG, blk, 0, stream>>>(xA, yb, p(16 + 6), p(16 + 7), xB);

    // ---- FFN ----
    gemm_kernel<1><<<g2048, blk, 0, stream>>>(xB, p(24), p(25), hb, (int)AROW, 4 * HIDN, HIDN);
    gemm_kernel<0><<<g512,  blk, 0, stream>>>(hb, p(26), p(27), yb, (int)AROW, HIDN, 4 * HIDN);
    float* dst = (L == NLAYERS - 1) ? (float*)d_out : xA;
    add_ln_kernel<<<lnG, blk, 0, stream>>>(xB, yb, p(28), p(29), dst);
    x = xA;
  }
  (void)in_sizes; (void)n_in; (void)out_size; (void)ws_size;
}
